// QuantLinearA8W8_57174604644899
// MI455X (gfx1250) — 8-bit matrix operand order repaired and checked by reading, compiled, not run on hardware
//
#include <hip/hip_runtime.h>
#include <hip/hip_bf16.h>

typedef int   v8i __attribute__((ext_vector_type(8)));
typedef int   v4i __attribute__((ext_vector_type(4)));
typedef int   v2i __attribute__((ext_vector_type(2)));
typedef float v4f __attribute__((ext_vector_type(4)));

#define M_DIM 4096
#define K_DIM 4096
#define N_DIM 11008
#define KP2   (K_DIM / 2)

#define TILE_M 128
#define TILE_N 128
#define TILE_K 64
#define LSTR   80
#define NCHUNK (K_DIM / TILE_K)

__global__ __launch_bounds__(256) void quant_act_kernel(
    const float* __restrict__ x, const float* __restrict__ clampv,
    signed char* __restrict__ qx, int* __restrict__ rowsum)
{
    __shared__ int red[256];
    const int row = blockIdx.x;
    const int t   = threadIdx.x;
    const float cv    = clampv[0];
    const float inv_s = 127.0f / cv;

    const float* xr   = x + (size_t)row * K_DIM;
    int*         qout = (int*)(qx + (size_t)row * K_DIM);

    int lsum = 0;
#pragma unroll
    for (int i = 0; i < 4; ++i) {
        const int dw = t + i * 256;
        v4f v = ((const v4f*)xr)[dw];
        int packed = 0;
#pragma unroll
        for (int j = 0; j < 4; ++j) {
            int qi = __float2int_rn(v[j] * inv_s);
            qi = qi > 127 ? 127 : (qi < -127 ? -127 : qi);
            lsum += qi;
            packed |= (qi & 0xFF) << (8 * j);
        }
        qout[dw] = packed;
    }

    red[t] = lsum;
    __syncthreads();
    for (int off = 128; off > 0; off >>= 1) {
        if (t < off) red[t] += red[t + off];
        __syncthreads();
    }
    if (t == 0) rowsum[row] = red[0];
}

__global__ __launch_bounds__(256) void unpack_w_kernel(
    const int* __restrict__ qweight, unsigned char* __restrict__ wu8)
{
    const size_t f = ((size_t)blockIdx.x * 256 + threadIdx.x) * 4;
    v4i q = *(const v4i*)(qweight + f);
    unsigned int p[2];
#pragma unroll
    for (int b = 0; b < 2; ++b) {
        const unsigned int b0 = (unsigned int)q[2 * b] & 0xFFu;
        const unsigned int b1 = (unsigned int)q[2 * b + 1] & 0xFFu;
        p[b] = ((b0 >> 4) & 15u) | ((b0 & 15u) << 8) |
               (((b1 >> 4) & 15u) << 16) | ((b1 & 15u) << 24);
    }
    v2i pv; pv.x = (int)p[0]; pv.y = (int)p[1];
    *(v2i*)(wu8 + 2 * f) = pv;
}

template <bool PRE>
__global__ __launch_bounds__(256) void w4a8_gemm_kernel(
    const signed char* __restrict__ qx, const int* __restrict__ rowsum,
    const int* __restrict__ qweight, const unsigned char* __restrict__ wu8,
    const int* __restrict__ qzeros, const float* __restrict__ scales,
    const float* __restrict__ bias, const float* __restrict__ clampv,
    float* __restrict__ out)
{
    __shared__ __align__(16) signed char   lA[2][TILE_M * LSTR];
    __shared__ __align__(16) unsigned char lB[2][TILE_N * LSTR];

    const int tid   = threadIdx.x;
    const int lane  = tid & 31;
    const int wv    = tid >> 5;
    const int w_m   = wv & 3;
    const int w_n   = wv >> 2;
    const int l16   = lane & 15;
    const int lhalf = lane >> 4;
    const int bm    = blockIdx.y * TILE_M;
    const int bn    = blockIdx.x * TILE_N;

    const int ar = tid >> 2, ac = (tid & 3) * 16;
    const int pr = tid >> 3, pj = (tid & 7) * 4;

    const float cv      = clampv[0];
    const float a_scale = cv / 127.0f;

    v8i acc[2][4];
#pragma unroll
    for (int mi = 0; mi < 2; ++mi)
#pragma unroll
        for (int ni = 0; ni < 4; ++ni)
            acc[mi][ni] = (v8i){0, 0, 0, 0, 0, 0, 0, 0};

    v4i aReg[2];
    v4i bReg[4];

    auto load_chunk = [&](int kk) {
#pragma unroll
        for (int i = 0; i < 2; ++i)
            aReg[i] = *(const v4i*)(qx + (size_t)(bm + ar + 64 * i) * K_DIM + kk + ac);
        if (PRE) {
#pragma unroll
            for (int i = 0; i < 2; ++i)
                bReg[i] = *(const v4i*)(wu8 + (size_t)(bn + ar + 64 * i) * K_DIM + kk + ac);
        } else {
#pragma unroll
            for (int i = 0; i < 4; ++i)
                bReg[i] = *(const v4i*)(qweight + (size_t)(bn + pr + 32 * i) * KP2 + (kk >> 1) + pj);
        }
    };

    auto store_chunk = [&](int buf) {
#pragma unroll
        for (int i = 0; i < 2; ++i)
            *(v4i*)(lA[buf] + (ar + 64 * i) * LSTR + ac) = aReg[i];
        if (PRE) {
#pragma unroll
            for (int i = 0; i < 2; ++i)
                *(v4i*)(lB[buf] + (ar + 64 * i) * LSTR + ac) = bReg[i];
        } else {
#pragma unroll
            for (int i = 0; i < 4; ++i) {
                unsigned int p[2];
#pragma unroll
                for (int b = 0; b < 2; ++b) {
                    const unsigned int b0 = (unsigned int)bReg[i][2 * b] & 0xFFu;
                    const unsigned int b1 = (unsigned int)bReg[i][2 * b + 1] & 0xFFu;
                    p[b] = ((b0 >> 4) & 15u) | ((b0 & 15u) << 8) |
                           (((b1 >> 4) & 15u) << 16) | ((b1 & 15u) << 24);
                }
                v2i pv; pv.x = (int)p[0]; pv.y = (int)p[1];
                *(v2i*)(lB[buf] + (pr + 32 * i) * LSTR + 2 * pj) = pv;
            }
        }
    };

    auto compute = [&](int buf) {
        v8i afr[2];
#pragma unroll
        for (int mi = 0; mi < 2; ++mi) {
            const signed char* ap =
                lA[buf] + (w_m * 32 + mi * 16 + l16) * LSTR + lhalf * 16;
            v2i a0 = *(const v2i*)(ap + 0);
            v2i a1 = *(const v2i*)(ap + 8);
            v2i a2 = *(const v2i*)(ap + 32);
            v2i a3 = *(const v2i*)(ap + 40);
            v8i a = {a0.x, a0.y, a1.x, a1.y, a2.x, a2.y, a3.x, a3.y};
            afr[mi] = a;
        }
        v8i bfr[4];
#pragma unroll
        for (int ni = 0; ni < 4; ++ni) {
            const unsigned char* bp =
                lB[buf] + (w_n * 64 + ni * 16 + l16) * LSTR + lhalf * 16;
            v4i b0 = *(const v4i*)(bp + 0);
            v4i b1 = *(const v4i*)(bp + 32);
            v8i b = {b0.x, b0.y, b0.z, b0.w, b1.x, b1.y, b1.z, b1.w};
            bfr[ni] = b;
        }
#pragma unroll
        for (int ni = 0; ni < 4; ++ni)
#pragma unroll
            for (int mi = 0; mi < 2; ++mi)
                acc[mi][ni] = __builtin_amdgcn_wmma_i32_16x16x64_iu8(
                    true, afr[mi], false, bfr[ni], acc[mi][ni], false, false);
    };

    load_chunk(0);
    store_chunk(0);

    for (int c2 = 0; c2 < NCHUNK; c2 += 2) {
        __syncthreads();
        load_chunk((c2 + 1) * TILE_K);
        compute(0);
        store_chunk(1);

        __syncthreads();
        const bool more = (c2 + 2) < NCHUNK;
        if (more) load_chunk((c2 + 2) * TILE_K);
        compute(1);
        if (more) store_chunk(0);
    }

#pragma unroll
    for (int mi = 0; mi < 2; ++mi) {
        const int mrow0 = bm + w_m * 32 + mi * 16 + 8 * lhalf;
        float rs[8];
#pragma unroll
        for (int r = 0; r < 8; ++r) rs[r] = (float)rowsum[mrow0 + r];
#pragma unroll
        for (int ni = 0; ni < 4; ++ni) {
            const int n   = bn + w_n * 64 + ni * 16 + l16;
            const float scl = scales[n] * a_scale;
            const float zf  = (float)qzeros[n];
            const float bi  = bias[n];
#pragma unroll
            for (int r = 0; r < 8; ++r) {
                const float o = scl * ((float)acc[mi][ni][r] - zf * rs[r]) + bi;
                out[(size_t)(mrow0 + r) * N_DIM + n] = o;
            }
        }
    }
}

extern "C" void kernel_launch(void* const* d_in, const int* in_sizes, int n_in,
                              void* d_out, int out_size, void* d_ws, size_t ws_size,
                              hipStream_t stream)
{
    const float* x       = (const float*)d_in[0];
    const int*   qweight = (const int*)d_in[1];
    const int*   qzeros  = (const int*)d_in[2];
    const float* scales  = (const float*)d_in[3];
    const float* bias    = (const float*)d_in[4];
    const float* clampv  = (const float*)d_in[5];
    float*       out     = (float*)d_out;

    const size_t rs_off = (size_t)M_DIM * K_DIM;
    const size_t wu_off = rs_off + (size_t)M_DIM * sizeof(int);
    const size_t need   = wu_off + (size_t)N_DIM * K_DIM;

    signed char* qx     = (signed char*)d_ws;
    int*         rowsum = (int*)((char*)d_ws + rs_off);

    quant_act_kernel<<<M_DIM, 256, 0, stream>>>(x, clampv, qx, rowsum);

    dim3 grid(N_DIM / TILE_N, M_DIM / TILE_M);
    if (ws_size >= need) {
        unsigned char* wu8 = (unsigned char*)d_ws + wu_off;
        const int nblk = (N_DIM * KP2 / 4) / 256;
        unpack_w_kernel<<<nblk, 256, 0, stream>>>(qweight, wu8);
        w4a8_gemm_kernel<true><<<grid, 256, 0, stream>>>(
            qx, rowsum, qweight, wu8, qzeros, scales, bias, clampv, out);
    } else {
        w4a8_gemm_kernel<false><<<grid, 256, 0, stream>>>(
            qx, rowsum, qweight, (const unsigned char*)nullptr,
            qzeros, scales, bias, clampv, out);
    }
}
